// SS2D_Substitute_v2_40037685133753
// MI455X (gfx1250) — compile-verified, matrix operand order checked by reading, not run on hardware
//
#include <hip/hip_runtime.h>
#include <hip/hip_bf16.h>

// ---------------- constants (from setup_inputs) ----------------
static constexpr int Bn  = 8;      // batch
static constexpr int Cd  = 128;    // d_model == inner
static constexpr int Hh  = 96, Ww = 96;
static constexpr int HW  = Hh * Ww;     // 9216
static constexpr int Lm  = HW;          // mamba sequence length
static constexpr int Gm  = 16;          // 2*B sequences per mamba (fwd+rev)
static constexpr int XP  = 48;          // padded R+2N (40 -> 48)

#define DEV __device__ __forceinline__

typedef __attribute__((ext_vector_type(16))) __bf16 v16bf;
typedef __attribute__((ext_vector_type(8)))  float  v8f;

union V16u { v16bf v; unsigned short s[16]; };

DEV unsigned short f2bf(float f) {
    unsigned int u = __float_as_uint(f);
    u += 0x7FFFu + ((u >> 16) & 1u);      // round-to-nearest-even
    return (unsigned short)(u >> 16);
}
DEV float siluf(float x)     { return x / (1.f + __expf(-x)); }
DEV float softplusf(float x) { return (x > 20.f) ? x : log1pf(__expf(x)); }

DEV v8f wmma_bf16(const V16u& a, const V16u& b, v8f c) {
    return __builtin_amdgcn_wmma_f32_16x16x32_bf16(false, a.v, false, b.v,
                                                   (short)0, c, false, false);
}

// ---------------- generic bf16 WMMA GEMM, 2x2 tile blocking ----------------
// C[batch][m][n] = sum_k A[m][k] * B[batch][k][n]
// A: bf16 row-major (lda = K).
// B: bf16, element at Bbase + batch*b_batch + n*bn + k   (k ALWAYS contiguous).
// C: f32 at Cbase + batch*c_batch + m*c_row + n.
// Each wave computes a 32x32 block (2x2 WMMA tiles); second M tile guarded
// wave-uniformly (EXEC stays all-ones around every WMMA).
__global__ void gemm_bf16_kernel(const unsigned short* __restrict__ A, int lda,
                                 const unsigned short* __restrict__ Bm,
                                 long b_batch, long bn,
                                 float* __restrict__ C, long c_batch, long c_row,
                                 int K, int Mtiles, int blocksM, int blocksN,
                                 int totalBlocks)
{
    int wave = blockIdx.x * (blockDim.x >> 5) + (threadIdx.x >> 5);
    if (wave >= totalBlocks) return;                 // wave-uniform
    int lane = threadIdx.x & 31;
    int half = lane >> 4, idx = lane & 15;
    int per   = blocksM * blocksN;
    int batch = wave / per;
    int rem   = wave - batch * per;
    int bm = rem / blocksN, bnn = rem - bm * blocksN;
    int m0 = bm * 32, n0 = bnn * 32;
    bool m1ok = (2 * bm + 1) < Mtiles;               // wave-uniform
    const unsigned short* Bb = Bm + (long)batch * b_batch;

    v8f acc00 = {}, acc01 = {}, acc10 = {}, acc11 = {};
    for (int k0 = 0; k0 < K; k0 += 32) {
        int kb = k0 + half * 16;
        V16u a0, a1, b0, b1;
        const unsigned short* ap0 = A + (long)(m0 + idx) * lda + kb;
        const unsigned short* ap1 = A + (long)(m1ok ? (m0 + 16 + idx) : (m0 + idx)) * lda + kb;
        #pragma unroll
        for (int i = 0; i < 16; ++i) a0.s[i] = ap0[i];
        #pragma unroll
        for (int i = 0; i < 16; ++i) a1.s[i] = ap1[i];
        const unsigned short* bp0 = Bb + (long)(n0 + idx) * bn + kb;
        const unsigned short* bp1 = Bb + (long)(n0 + 16 + idx) * bn + kb;
        #pragma unroll
        for (int i = 0; i < 16; ++i) b0.s[i] = bp0[i];
        #pragma unroll
        for (int i = 0; i < 16; ++i) b1.s[i] = bp1[i];
        acc00 = wmma_bf16(a0, b0, acc00);
        acc01 = wmma_bf16(a0, b1, acc01);
        if (m1ok) {
            acc10 = wmma_bf16(a1, b0, acc10);
            acc11 = wmma_bf16(a1, b1, acc11);
        }
    }
    float* Cb = C + (long)batch * c_batch;
    #pragma unroll
    for (int v = 0; v < 8; ++v) {
        int r = m0 + v + half * 8;
        Cb[(long)r * c_row + n0 + idx]      = acc00[v];
        Cb[(long)r * c_row + n0 + 16 + idx] = acc01[v];
    }
    if (m1ok) {
        #pragma unroll
        for (int v = 0; v < 8; ++v) {
            int r = m0 + 16 + v + half * 8;
            Cb[(long)r * c_row + n0 + idx]      = acc10[v];
            Cb[(long)r * c_row + n0 + 16 + idx] = acc11[v];
        }
    }
}

// ---------------- tiled transpose + f32->bf16 convert ----------------
// src: f32 (batch, R, C) channel-major;  dst: bf16 (batch, C, R) position-major.
// R, C multiples of 32.  256 threads = 32x8.
__global__ void transpose_cvt_kernel(const float* __restrict__ src,
                                     unsigned short* __restrict__ dst,
                                     int R, int Ccols) {
    __shared__ float tile[32][33];
    int rtiles = R >> 5, ctiles = Ccols >> 5;
    int per = rtiles * ctiles;
    int batch = blockIdx.x / per;
    int rem   = blockIdx.x % per;
    int rt = rem / ctiles, ct = rem % ctiles;
    const float* s = src + (long)batch * R * Ccols;
    unsigned short* d = dst + (long)batch * R * Ccols;
    int tx = threadIdx.x & 31, ty = threadIdx.x >> 5;
    int r0 = rt * 32, c0 = ct * 32;
    #pragma unroll
    for (int j = 0; j < 4; ++j)
        tile[ty + 8 * j][tx] = s[(long)(r0 + ty + 8 * j) * Ccols + c0 + tx];
    __syncthreads();
    #pragma unroll
    for (int j = 0; j < 4; ++j)
        d[(long)(c0 + ty + 8 * j) * R + r0 + tx] = f2bf(tile[tx][ty + 8 * j]);
}

// ---------------- elementwise / helper kernels ----------------
__global__ void cvt_bf16_kernel(const float* __restrict__ s, unsigned short* __restrict__ d, long n) {
    long i = (long)blockIdx.x * 256 + threadIdx.x;
    if (i < n) d[i] = f2bf(s[i]);
}

// 3x3 depthwise SAME conv on x-branch of xz (B,256,HW): channels 0..127 -> xc (B,128,HW)
__global__ void dwconv_kernel(const float* __restrict__ xz, const float* __restrict__ w,
                              float* __restrict__ out) {
    long i = (long)blockIdx.x * 256 + threadIdx.x;
    long total = (long)Bn * Cd * HW; if (i >= total) return;
    int hw = (int)(i % HW); long bc = i / HW;
    int c = (int)(bc % Cd), b = (int)(bc / Cd);
    int h = hw / Ww, wx = hw % Ww;
    const float* src = xz + ((long)b * 256 + c) * HW;
    const float* k   = w + c * 9;
    float a = 0.f;
    #pragma unroll
    for (int dy = 0; dy < 3; ++dy) {
        int hh = h + dy - 1; if ((unsigned)hh >= (unsigned)Hh) continue;
        #pragma unroll
        for (int dx = 0; dx < 3; ++dx) {
            int ww2 = wx + dx - 1; if ((unsigned)ww2 >= (unsigned)Ww) continue;
            a += src[hh * Ww + ww2] * k[dy * 3 + dx];
        }
    }
    out[i] = a;
}

// mean over HW of silu(z-branch)
__global__ void gate_mean_kernel(const float* __restrict__ xz, float* __restrict__ gm) {
    int bc = blockIdx.x; int b = bc >> 7, c = bc & 127;
    const float* z = xz + ((long)b * 256 + 128 + c) * HW;
    float s = 0.f;
    for (int i = threadIdx.x; i < HW; i += 256) s += siluf(z[i]);
    __shared__ float red[256];
    red[threadIdx.x] = s; __syncthreads();
    for (int st = 128; st > 0; st >>= 1) {
        if (threadIdx.x < st) red[threadIdx.x] += red[threadIdx.x + st];
        __syncthreads();
    }
    if (threadIdx.x == 0) gm[bc] = red[0] / (float)HW;
}

__global__ void gate_mlp1_kernel(const float* __restrict__ gm, const float* __restrict__ w1,
                                 const float* __restrict__ b1, float* __restrict__ hid) {
    int t = threadIdx.x; int b = t >> 5, j = t & 31;
    float a = b1[j];
    const float* g = gm + b * 128; const float* w = w1 + j * 128;
    for (int c = 0; c < 128; ++c) a += g[c] * w[c];
    hid[t] = siluf(a);
}

__global__ void gate_mlp2_kernel(const float* __restrict__ hid, const float* __restrict__ w2,
                                 const float* __restrict__ b2, float* __restrict__ gv) {
    int t = blockIdx.x * 256 + threadIdx.x; if (t >= 1024) return;
    int b = t >> 7, c = t & 127;
    float a = b2[c];
    const float* h = hid + b * 32; const float* w = w2 + c * 32;
    for (int j = 0; j < 32; ++j) a += h[j] * w[j];
    gv[t] = 1.f / (1.f + __expf(-a));
}

// build (16,L,128) bf16 sequence tensor (fwd + reversed) from xc (B,128,H,W)
__global__ void build_seq_kernel(const float* __restrict__ xc, unsigned short* __restrict__ seq,
                                 int vert) {
    long i = (long)blockIdx.x * 256 + threadIdx.x;
    long total = (long)Gm * Lm * Cd; if (i >= total) return;
    int d = (int)(i & 127); long gl = i >> 7;
    int l = (int)(gl % Lm), g = (int)(gl / Lm);
    int b = g & 7, rev = g >> 3;
    int pos = rev ? (Lm - 1 - l) : l;
    int h, w;
    if (!vert) { h = pos / Ww; w = pos % Ww; }
    else       { w = pos / Hh; h = pos % Hh; }
    seq[i] = f2bf(xc[((long)b * Cd + d) * HW + h * Ww + w]);
}

// causal depthwise conv1d along L + silu on x-half of xzm (16,256,L) -> xcf (16,128,L)
__global__ void mconv_kernel(const float* __restrict__ xzm, const float* __restrict__ cw,
                             const float* __restrict__ cb, float* __restrict__ xcf, int K) {
    long i = (long)blockIdx.x * 256 + threadIdx.x;
    long total = (long)Gm * Cd * Lm; if (i >= total) return;
    int l = (int)(i % Lm); long go = i / Lm;
    int o = (int)(go % Cd), g = (int)(go / Cd);
    const float* src = xzm + ((long)g * 256 + o) * Lm;
    float a = cb[o];
    for (int k = 0; k < K; ++k) {
        int ls = l - (K - 1) + k;
        if (ls >= 0) a += src[ls] * cw[o * K + k];
    }
    xcf[i] = siluf(a);
}

// dt = softplus(x_dbl[:, :8] @ dt_w.T + dt_b)   (K=8: VALU, too small for WMMA)
__global__ void dt_kernel(const float* __restrict__ xdbl, const float* __restrict__ dtw,
                          const float* __restrict__ dtb, float* __restrict__ dtout) {
    long i = (long)blockIdx.x * 256 + threadIdx.x;
    long total = (long)Gm * Cd * Lm; if (i >= total) return;
    int l = (int)(i % Lm); long gd = i / Lm;
    int d = (int)(gd % Cd), g = (int)(gd / Cd);
    const float* xd = xdbl + (long)g * XP * Lm + l;
    float a = dtb[d];
    #pragma unroll
    for (int r = 0; r < 8; ++r) a += dtw[d * 8 + r] * xd[(long)r * Lm];
    dtout[i] = softplusf(a);
}

// selective scan: 1 block per sequence g, thread d holds h[0..15] in VGPRs.
// Output ym written POS-MAJOR (16,L,128) bf16: coalesced stores + contiguous
// k-runs for the following WMMA out-projection.
__global__ void scan_kernel(const float* __restrict__ dtp, const float* __restrict__ up,
                            const float* __restrict__ xzm, const float* __restrict__ xdbl,
                            const float* __restrict__ Alog, const float* __restrict__ Dp,
                            unsigned short* __restrict__ ym) {
    const int T = 32;
    __shared__ float sBC[T * 32];
    __shared__ float sDT[T * 128];
    __shared__ float sU [T * 128];
    __shared__ float sZ [T * 128];
    int g = blockIdx.x, d = threadIdx.x;
    const float* dt_g = dtp + (long)g * Cd * Lm;
    const float* u_g  = up  + (long)g * Cd * Lm;
    const float* z_g  = xzm + ((long)g * 256 + 128) * Lm;
    const float* xd_g = xdbl + (long)g * XP * Lm;
    unsigned short* ym_g = ym + (long)g * Lm * Cd;

    float A_row[16], h[16];
    #pragma unroll
    for (int n = 0; n < 16; ++n) { A_row[n] = -__expf(Alog[d * 16 + n]); h[n] = 0.f; }
    float Dd = Dp[d];

    for (int l0 = 0; l0 < Lm; l0 += T) {
        for (int i = d; i < 32 * T; i += 128) {
            int j = i / T, t = i % T;
            sBC[t * 32 + j] = xd_g[(long)(8 + j) * Lm + l0 + t];
        }
        for (int i = d; i < 128 * T; i += 128) {
            int dd = i / T, t = i % T;
            sDT[t * 128 + dd] = dt_g[(long)dd * Lm + l0 + t];
            sU [t * 128 + dd] = u_g [(long)dd * Lm + l0 + t];
            sZ [t * 128 + dd] = z_g [(long)dd * Lm + l0 + t];
        }
        __syncthreads();
        for (int t = 0; t < T; ++t) {
            float dtv = sDT[t * 128 + d];
            float u   = sU [t * 128 + d];
            float zz  = sZ [t * 128 + d];
            float du = dtv * u, y = 0.f;
            #pragma unroll
            for (int n = 0; n < 16; ++n) {
                h[n] = __expf(dtv * A_row[n]) * h[n] + du * sBC[t * 32 + n];
                y += h[n] * sBC[t * 32 + 16 + n];
            }
            ym_g[(long)(l0 + t) * Cd + d] = f2bf((y + u * Dd) * siluf(zz));
        }
        __syncthreads();
    }
}

// scatter out-proj result (16,128,L) into ycat (B,HW,512) bf16 with direction remap
__global__ void scatter_kernel(const float* __restrict__ yo, unsigned short* __restrict__ ycat,
                               int vert) {
    long i = (long)blockIdx.x * 256 + threadIdx.x;
    long total = (long)Gm * Cd * Lm; if (i >= total) return;
    int l = (int)(i % Lm); long go = i / Lm;
    int o = (int)(go % Cd), g = (int)(go / Cd);
    int b = g & 7, rev = g >> 3;
    int pos = rev ? (Lm - 1 - l) : l;
    int hw, off;
    if (!vert) { hw = pos; off = rev ? 128 : 0; }
    else { int w = pos / Hh, h = pos % Hh; hw = h * Ww + w; off = rev ? 384 : 256; }
    ycat[((long)b * HW + hw) * 512 + off + o] = f2bf(yo[i]);
}

// y * silu(z) * gate -> f32 channel-major (transposed to bf16 pos-major afterwards)
__global__ void gmul_kernel(const float* __restrict__ yf, const float* __restrict__ xz,
                            const float* __restrict__ gv, float* __restrict__ ygf) {
    long i = (long)blockIdx.x * 256 + threadIdx.x;
    long total = (long)Bn * Cd * HW; if (i >= total) return;
    int hw = (int)(i % HW); long bc = i / HW;
    int c = (int)(bc % Cd), b = (int)(bc / Cd);
    float z = xz[((long)b * 256 + 128 + c) * HW + hw];
    ygf[i] = yf[i] * siluf(z) * gv[b * 128 + c];
}

// ---------------- host launcher ----------------
struct MParams {
    const float *in_w, *conv_w, *conv_b, *xproj_w, *dt_w, *dt_b, *A_log, *D, *out_w;
};

extern "C" void kernel_launch(void* const* d_in, const int* in_sizes, int n_in,
                              void* d_out, int out_size, void* d_ws, size_t ws_size,
                              hipStream_t stream) {
    (void)n_in; (void)out_size; (void)ws_size;
    const float* x      = (const float*)d_in[0];
    const float* in_w   = (const float*)d_in[1];
    const float* dw_w   = (const float*)d_in[2];
    const float* fuse_w = (const float*)d_in[3];
    const float* g_w1   = (const float*)d_in[4];
    const float* g_b1   = (const float*)d_in[5];
    const float* g_w2   = (const float*)d_in[6];
    const float* g_b2   = (const float*)d_in[7];
    const float* out_w  = (const float*)d_in[8];
    MParams pm[2];
    for (int m = 0; m < 2; ++m) {
        int o = 9 + m * 9;
        pm[m] = { (const float*)d_in[o+0], (const float*)d_in[o+1], (const float*)d_in[o+2],
                  (const float*)d_in[o+3], (const float*)d_in[o+4], (const float*)d_in[o+5],
                  (const float*)d_in[o+6], (const float*)d_in[o+7], (const float*)d_in[o+8] };
    }
    const int Kc = in_sizes[10] / Cd;   // mamba conv kernel width (=2)

    // ---- workspace layout ----
    char* wsp = (char*)d_ws; size_t off = 0;
    auto alloc = [&](size_t bytes) -> void* {
        void* p = wsp + off; off += (bytes + 255) & ~(size_t)255; return p;
    };
    unsigned short* wb_in   = (unsigned short*)alloc((size_t)256 * 128 * 2);
    unsigned short* wb_fuse = (unsigned short*)alloc((size_t)128 * 512 * 2);
    unsigned short* wb_outO = (unsigned short*)alloc((size_t)128 * 128 * 2);
    unsigned short* wb_m_in = (unsigned short*)alloc((size_t)256 * 128 * 2);
    unsigned short* wb_m_xp = (unsigned short*)alloc((size_t)XP * 128 * 2);
    unsigned short* wb_m_out= (unsigned short*)alloc((size_t)128 * 128 * 2);
    unsigned short* xbf  = (unsigned short*)alloc((size_t)Bn * HW * Cd * 2);   // (B,HW,C) bf16
    float*          xz   = (float*)alloc((size_t)Bn * 256 * HW * 4);           // (B,256,HW)
    float*          xcsp = (float*)alloc((size_t)Bn * Cd * HW * 4);            // (B,128,HW)
    float*          gmv  = (float*)alloc(1024 * 4);
    float*          hid  = (float*)alloc(256 * 4);
    float*          gv   = (float*)alloc(1024 * 4);
    unsigned short* seq  = (unsigned short*)alloc((size_t)Gm * Lm * Cd * 2);   // (16,L,128) bf16
    float*          xzm  = (float*)alloc((size_t)Gm * 256 * Lm * 4);           // (16,256,L)
    float*          xcf  = (float*)alloc((size_t)Gm * Cd * Lm * 4);            // (16,128,L)
    unsigned short* xcb  = (unsigned short*)alloc((size_t)Gm * Lm * Cd * 2);   // (16,L,128) bf16
    float*          xdbl = (float*)alloc((size_t)Gm * XP * Lm * 4);            // (16,48,L)
    float*          dtb  = (float*)alloc((size_t)Gm * Cd * Lm * 4);            // (16,128,L)
    unsigned short* ymb  = (unsigned short*)alloc((size_t)Gm * Lm * Cd * 2);   // (16,L,128) bf16
    float*          yo   = (float*)alloc((size_t)Gm * Cd * Lm * 4);            // (16,128,L) / reused
    unsigned short* ycat = (unsigned short*)alloc((size_t)Bn * HW * 512 * 2);  // (B,HW,512) bf16
    float*          yf   = (float*)alloc((size_t)Bn * Cd * HW * 4);            // (B,128,HW)
    unsigned short* yg   = (unsigned short*)alloc((size_t)Bn * HW * Cd * 2);   // (B,HW,128) bf16
    float*          ygf  = yo;  // reuse (free after last scatter)

    auto cvt = [&](const float* s, unsigned short* d, long n) {
        cvt_bf16_kernel<<<(int)((n + 255) / 256), 256, 0, stream>>>(s, d, n);
    };
    auto tcvt = [&](const float* s, unsigned short* d, int R, int Ccols, int batches) {
        int blocks = batches * (R >> 5) * (Ccols >> 5);
        transpose_cvt_kernel<<<blocks, 256, 0, stream>>>(s, d, R, Ccols);
    };
    auto gemm = [&](const unsigned short* A, int lda, const unsigned short* Bm,
                    long b_batch, long bn, float* C, long c_batch, long c_row,
                    int K, int M, int Ncol, int batches) {
        int Mtiles = M / 16;
        int blocksM = (Mtiles + 1) / 2, blocksN = Ncol / 32;
        int total = batches * blocksM * blocksN;
        gemm_bf16_kernel<<<(total + 7) / 8, 256, 0, stream>>>(
            A, lda, Bm, b_batch, bn, C, c_batch, c_row, K, Mtiles, blocksM, blocksN, total);
    };

    // ---- static weights ----
    cvt(in_w,   wb_in,   256 * 128);
    cvt(fuse_w, wb_fuse, 128 * 512);
    cvt(out_w,  wb_outO, 128 * 128);

    // ---- x -> bf16 pos-major, then outer in-proj ----
    tcvt(x, xbf, Cd, HW, Bn);
    gemm(wb_in, 128, xbf, (long)HW * Cd, Cd, xz, (long)256 * HW, HW, 128, 256, HW, Bn);

    // ---- 3x3 depthwise conv on x-branch ----
    {
        long n = (long)Bn * Cd * HW;
        dwconv_kernel<<<(int)((n + 255) / 256), 256, 0, stream>>>(xz, dw_w, xcsp);
    }

    // ---- channel gate ----
    gate_mean_kernel<<<Bn * Cd, 256, 0, stream>>>(xz, gmv);
    gate_mlp1_kernel<<<1, 256, 0, stream>>>(gmv, g_w1, g_b1, hid);
    gate_mlp2_kernel<<<4, 256, 0, stream>>>(hid, g_w2, g_b2, gv);

    // ---- two mamba passes (horizontal then vertical), fwd+rev batched ----
    for (int m = 0; m < 2; ++m) {
        const MParams& p = pm[m];
        const int vert = m;
        cvt(p.in_w, wb_m_in, 256 * 128);
        hipMemsetAsync(wb_m_xp, 0, (size_t)XP * 128 * 2, stream);
        cvt(p.xproj_w, wb_m_xp, 40 * 128);
        cvt(p.out_w, wb_m_out, 128 * 128);

        long nGL = (long)Gm * Lm * Cd;
        build_seq_kernel<<<(int)((nGL + 255) / 256), 256, 0, stream>>>(xcsp, seq, vert);
        // in-proj: xzm[g,o,l] = sum_d in_w[o,d]*seq[g,l,d]
        gemm(wb_m_in, 128, seq, (long)Lm * Cd, Cd, xzm, (long)256 * Lm, Lm, 128, 256, Lm, Gm);
        // causal conv1d + silu (channel-major for the scan)
        mconv_kernel<<<(int)((nGL + 255) / 256), 256, 0, stream>>>(xzm, p.conv_w, p.conv_b,
                                                                   xcf, Kc);
        // bf16 pos-major copy for xproj GEMM
        tcvt(xcf, xcb, Cd, Lm, Gm);
        // xproj: x_dbl[g,j,l] = sum_d xproj_w[j,d]*xc[g,l,d]
        gemm(wb_m_xp, 128, xcb, (long)Lm * Cd, Cd, xdbl, (long)XP * Lm, Lm, 128, XP, Lm, Gm);
        // dt projection + softplus
        dt_kernel<<<(int)((nGL + 255) / 256), 256, 0, stream>>>(xdbl, p.dt_w, p.dt_b, dtb);
        // selective scan + D skip + silu(z) gating -> bf16 pos-major
        scan_kernel<<<Gm, 128, 0, stream>>>(dtb, xcf, xzm, xdbl, p.A_log, p.D, ymb);
        // out-proj: yo[g,o,l] = sum_d out_w[o,d]*ym[g,l,d]
        gemm(wb_m_out, 128, ymb, (long)Lm * Cd, Cd, yo, (long)Cd * Lm, Lm, 128, 128, Lm, Gm);
        // scatter into ycat with direction/reversal remap
        scatter_kernel<<<(int)((nGL + 255) / 256), 256, 0, stream>>>(yo, ycat, vert);
    }

    // ---- fuse: yf[b,c,hw] = sum_j fuse_w[c,j]*ycat[b,hw,j]  (K=512) ----
    gemm(wb_fuse, 512, ycat, (long)HW * 512, 512, yf, (long)Cd * HW, HW, 512, 128, HW, Bn);

    // ---- gating multiply (f32 channel-major), then bf16 pos-major ----
    {
        long n = (long)Bn * Cd * HW;
        gmul_kernel<<<(int)((n + 255) / 256), 256, 0, stream>>>(yf, xz, gv, ygf);
    }
    tcvt(ygf, yg, Cd, HW, Bn);

    // ---- final 1x1 out conv -> d_out (B,128,H,W) f32 ----
    gemm(wb_outO, 128, yg, (long)HW * Cd, Cd, (float*)d_out, (long)Cd * HW, HW,
         128, 128, HW, Bn);
}
